// NonLinearRandomProjectionAttention_53678501265448
// MI455X (gfx1250) — compile-verified, matrix operand order checked by reading, not run on hardware
//
#include <hip/hip_runtime.h>
#include <hip/hip_bf16.h>

// ---------------------------------------------------------------------------
// NonLinearRandomProjectionAttention for MI455X (gfx1250, wave32, WMMA)
//
//   P   = mean_p Z[p]              [1024,32]
//   xp  = x @ P, yp = y @ P        [4096,32]
//   A   = xp @ yp^T                [4096,4096] fp32 via v_wmma_f32_16x16x4_f32
//   out = softmax(A) @ v           [4096,1024] bf16 in / fp32 acc via
//                                              v_wmma_f32_16x16x32_bf16
//   V is pre-packed once into the exact per-lane WMMA B-fragment layout so
//   the hot PV loop does only: stage probs (LDS) -> gather A -> 2x b128 -> wmma.
// ---------------------------------------------------------------------------

typedef float        v2f   __attribute__((ext_vector_type(2)));
typedef float        v8f   __attribute__((ext_vector_type(8)));
typedef __bf16       v16bf __attribute__((ext_vector_type(16)));
typedef unsigned int v4u   __attribute__((ext_vector_type(4)));

#define N_ROWS 4096
#define DV     1024
#define KIN    1024
#define ID     32      // inner dim
#define NPROJ  24

__device__ __forceinline__ unsigned short f32_to_bf16_rne(float f) {
    unsigned int u = __float_as_uint(f);
    unsigned int r = (u + 0x7FFFu + ((u >> 16) & 1u)) >> 16;
    return (unsigned short)r;
}

__device__ __forceinline__ int frag_kbase(int j) {   // even K base of dword j
    return (j < 4) ? (2 * j) : (16 + 2 * (j - 4));
}

// ---------------------------------------------------------------------------
// K1: P[i] = (1/24) * sum_p Z[p*32768 + i]
// ---------------------------------------------------------------------------
__global__ __launch_bounds__(256)
void nlrpa_reduceZ(const float* __restrict__ Z, float* __restrict__ P) {
    int i = blockIdx.x * 256 + threadIdx.x;
    float s = 0.f;
#pragma unroll
    for (int p = 0; p < NPROJ; ++p) s += Z[p * (KIN * ID) + i];
    P[i] = s * (1.0f / (float)NPROJ);
}

// ---------------------------------------------------------------------------
// K2: projections. One wave per row; lane c owns output column c.
// ---------------------------------------------------------------------------
__global__ __launch_bounds__(256)
void nlrpa_proj(const float* __restrict__ x, const float* __restrict__ y,
                const float* __restrict__ P,
                float* __restrict__ xp, float* __restrict__ yp) {
    int wid  = threadIdx.x >> 5;
    int lane = threadIdx.x & 31;
    int gw   = blockIdx.x * 8 + wid;
    const float* src; float* dst; int row;
    if (gw < N_ROWS) { src = x; dst = xp; row = gw; }
    else             { src = y; dst = yp; row = gw - N_ROWS; }
    const float* xr = src + (size_t)row * KIN;
    float acc = 0.f;
#pragma unroll 8
    for (int j = 0; j < KIN; ++j)
        acc = fmaf(xr[j], P[j * ID + lane], acc);
    dst[row * ID + lane] = acc;
}

// ---------------------------------------------------------------------------
// K3: pack V into bf16 WMMA B-fragment layout:
//     vfrag[((cc*64 + dt)*32 + lane)*8 + j] =
//        pack(bf16(v[cc*32 + kb][dt*16 + n]), bf16(v[cc*32 + kb + 1][...]))
//     kb = frag_kbase(j) + (lane>=16 ? 8 : 0), n = lane&15.
// ---------------------------------------------------------------------------
__global__ __launch_bounds__(256)
void nlrpa_packV(const float* __restrict__ v, unsigned int* __restrict__ vfrag) {
    int gid  = blockIdx.x * 256 + threadIdx.x;   // 0 .. 2^21-1
    int j    = gid & 7;
    int lane = (gid >> 3) & 31;
    int dt   = (gid >> 8) & 63;
    int cc   = gid >> 14;                        // 0..127
    int kb   = frag_kbase(j) + ((lane >> 4) << 3);
    int k    = cc * 32 + kb;
    int n    = dt * 16 + (lane & 15);
    unsigned int lo = f32_to_bf16_rne(v[(size_t)k * DV + n]);
    unsigned int hi = f32_to_bf16_rne(v[(size_t)(k + 1) * DV + n]);
    vfrag[gid] = lo | (hi << 16);
}

// ---------------------------------------------------------------------------
// K4: attention = xp @ yp^T, fp32 WMMA 16x16x4, one wave per 16x16 tile.
// ---------------------------------------------------------------------------
__global__ __launch_bounds__(256)
void nlrpa_attn(const float* __restrict__ xp, const float* __restrict__ yp,
                float* __restrict__ att) {
    int wid  = threadIdx.x >> 5;
    int lane = threadIdx.x & 31;
    int tile = blockIdx.x * 8 + wid;          // 0..65535
    int tr = tile >> 8, tc = tile & 255;
    int m   = lane & 15;
    int khi = (lane >> 4) << 1;               // 0 or 2

    const float* arow = xp + (size_t)(tr * 16 + m) * ID;
    const float* brow = yp + (size_t)(tc * 16 + m) * ID;

    v8f c = {};
#pragma unroll
    for (int kk = 0; kk < 8; ++kk) {
        int k = kk * 4 + khi;
        v2f a, b;
        a.x = arow[k]; a.y = arow[k + 1];
        b.x = brow[k]; b.y = brow[k + 1];
        c = __builtin_amdgcn_wmma_f32_16x16x4_f32(
                false, a, false, b, (short)0, c, false, false);
    }

    int r0 = tr * 16 + ((lane >> 4) << 3);
    int c0 = tc * 16 + m;
#pragma unroll
    for (int r = 0; r < 8; ++r)
        att[(size_t)(r0 + r) * N_ROWS + c0] = c[r];
}

// ---------------------------------------------------------------------------
// K5: per-row softmax stats: rowmax and 1/sum(exp(a - max)). One block/row.
// ---------------------------------------------------------------------------
__global__ __launch_bounds__(256)
void nlrpa_rowstat(const float* __restrict__ att,
                   float* __restrict__ rowmax, float* __restrict__ rowinv) {
    __shared__ float red[256];
    int row = blockIdx.x, tid = threadIdx.x;
    const float* ar = att + (size_t)row * N_ROWS;

    float mx = -3.402823466e38f;
    for (int t = tid; t < N_ROWS; t += 256) mx = fmaxf(mx, ar[t]);
    red[tid] = mx; __syncthreads();
    for (int s = 128; s > 0; s >>= 1) {
        if (tid < s) red[tid] = fmaxf(red[tid], red[tid + s]);
        __syncthreads();
    }
    float m = red[0]; __syncthreads();

    float sum = 0.f;
    for (int t = tid; t < N_ROWS; t += 256) sum += __expf(ar[t] - m);
    red[tid] = sum; __syncthreads();
    for (int s = 128; s > 0; s >>= 1) {
        if (tid < s) red[tid] += red[tid + s];
        __syncthreads();
    }
    if (tid == 0) { rowmax[row] = m; rowinv[row] = 1.0f / red[0]; }
}

// ---------------------------------------------------------------------------
// K6: out = softmax(A) @ v.
//     Block = 16 rows x 256 cols (8 waves x two 16-col d-tiles each).
//     K-step 64 per barrier; per K-chunk of 32: gather shared bf16 prob
//     fragment from LDS, four b128 loads (two per d-tile) of pre-packed V fragments, 2 WMMAs.
// ---------------------------------------------------------------------------
__global__ __launch_bounds__(256)
void nlrpa_pv(const float* __restrict__ att, const unsigned int* __restrict__ vfrag,
              const float* __restrict__ rowmax, const float* __restrict__ rowinv,
              float* __restrict__ outp) {
    __shared__ unsigned short sA[16 * 64];       // prob tile bf16, [m][k], k<64

    int tid  = threadIdx.x;
    int wid  = tid >> 5;
    int lane = tid & 31;
    int rowBlk = blockIdx.x >> 2;                // 256 row blocks
    int dGrp   = blockIdx.x & 3;                 // 4 groups of 256 cols
    int r0  = rowBlk * 16;
    int dt0 = dGrp * 16 + wid * 2;               // first of two d-tiles (16 cols each)
    int m   = lane & 15;
    int laneHiK = (lane >> 4) << 3;              // 0 or 8

    v8f acc0 = {}, acc1 = {};
    const unsigned int* a32 = (const unsigned int*)sA;

    for (int c0 = 0; c0 < N_ROWS; c0 += 64) {
        // stage 16x64 probability tile (4 values per thread, coalesced)
#pragma unroll
        for (int it = 0; it < 4; ++it) {
            int idx = tid + it * 256;
            int mm = idx >> 6, kk = idx & 63;
            int rr = r0 + mm;
            float a = att[(size_t)rr * N_ROWS + c0 + kk];
            float p = __expf(a - rowmax[rr]) * rowinv[rr];
            sA[idx] = f32_to_bf16_rne(p);
        }
        if (c0 + 64 < N_ROWS)
            __builtin_prefetch(&att[(size_t)(r0 + m) * N_ROWS + c0 + 64], 0, 1);
        __syncthreads();

#pragma unroll
        for (int q = 0; q < 2; ++q) {            // two K=32 chunks
            union { v16bf v; unsigned int u[8]; } A;
#pragma unroll
            for (int j = 0; j < 8; ++j) {
                int kb = frag_kbase(j) + laneHiK;
                A.u[j] = a32[m * 32 + q * 16 + (kb >> 1)];
            }
            int cc = (c0 >> 5) + q;
            const v4u* b0p = (const v4u*)(vfrag +
                ((size_t)(cc * 64 + dt0) * 32 + lane) * 8);
            const v4u* b1p = (const v4u*)(vfrag +
                ((size_t)(cc * 64 + dt0 + 1) * 32 + lane) * 8);
            union { v16bf v; v4u q2[2]; } B0, B1;
            B0.q2[0] = b0p[0]; B0.q2[1] = b0p[1];
            B1.q2[0] = b1p[0]; B1.q2[1] = b1p[1];
            acc0 = __builtin_amdgcn_wmma_f32_16x16x32_bf16(
                       false, A.v, false, B0.v, (short)0, acc0, false, false);
            acc1 = __builtin_amdgcn_wmma_f32_16x16x32_bf16(
                       false, A.v, false, B1.v, (short)0, acc1, false, false);
        }
        __syncthreads();
    }

    int rr = r0 + ((lane >> 4) << 3);
    int d0 = dt0 * 16 + m;
#pragma unroll
    for (int r = 0; r < 8; ++r) {
        outp[(size_t)(rr + r) * DV + d0]      = acc0[r];
        outp[(size_t)(rr + r) * DV + d0 + 16] = acc1[r];
    }
}

// ---------------------------------------------------------------------------
extern "C" void kernel_launch(void* const* d_in, const int* in_sizes, int n_in,
                              void* d_out, int out_size, void* d_ws, size_t ws_size,
                              hipStream_t stream) {
    const float* x = (const float*)d_in[0];
    const float* y = (const float*)d_in[1];
    const float* v = (const float*)d_in[2];
    const float* Z = (const float*)d_in[3];

    float* att  = (float*)d_out;                       // [4096,4096]
    float* outp = att + (size_t)N_ROWS * N_ROWS;       // [4096,1024]

    float* P      = (float*)d_ws;                      //  32768 f32
    float* xp     = P  + KIN * ID;                     // 131072 f32
    float* yp     = xp + N_ROWS * ID;                  // 131072 f32
    float* rowmax = yp + N_ROWS * ID;                  //   4096 f32
    float* rowinv = rowmax + N_ROWS;                   //   4096 f32
    unsigned int* vfrag = (unsigned int*)(rowinv + N_ROWS);  // 2^21 dwords (8 MB), 16B-aligned

    nlrpa_reduceZ<<<(KIN * ID) / 256, 256, 0, stream>>>(Z, P);
    nlrpa_packV  <<<(128 * 64 * 32 * 8) / 256, 256, 0, stream>>>(v, vfrag);
    nlrpa_proj   <<<(2 * N_ROWS) / 8, 256, 0, stream>>>(x, y, P, xp, yp);
    nlrpa_attn   <<<(N_ROWS / 16) * (N_ROWS / 16) / 8, 256, 0, stream>>>(xp, yp, att);
    nlrpa_rowstat<<<N_ROWS, 256, 0, stream>>>(att, rowmax, rowinv);
    nlrpa_pv     <<<(N_ROWS / 16) * (DV / 256), 256, 0, stream>>>(att, vfrag, rowmax, rowinv, outp);
}
